// GNN_node_Virtualnode_43980465111484
// MI455X (gfx1250) — matrix operand order repaired and checked by reading, not compiled, not run on hardware
//
#include <hip/hip_runtime.h>

// ---------------- problem constants (match reference) ----------------
#define NN      8192      // nodes
#define EE      131072    // edges
#define BB      64        // graphs
#define DD      256       // feature dim
#define NPG     128       // nodes per graph
#define VBSZ    8         // bond vocab
#define LL      2         // layers
#define NIT     5         // iterations per layer

#define LDS_W_BYTES (256 * 256 * 2)   // full bf16 weight matrix in LDS (128 KB)

typedef __bf16 bf16_t;
typedef __bf16 v16bf __attribute__((ext_vector_type(16)));
typedef __bf16 v8bf  __attribute__((ext_vector_type(8)));
typedef float  v8f   __attribute__((ext_vector_type(8)));

typedef unsigned int u32x4 __attribute__((ext_vector_type(4)));
typedef int          i32x4 __attribute__((ext_vector_type(4)));
typedef int          i32x8 __attribute__((ext_vector_type(8)));

#if __has_builtin(__builtin_amdgcn_tensor_load_to_lds)
#define HAVE_TDM 1
#else
#define HAVE_TDM 0
#endif

// ---------------- weight prep: W[k][n] f32 -> Wt[n][k] bf16 ----------------
__global__ __launch_bounds__(256)
void k_transpose_bf16(const float* __restrict__ W, bf16_t* __restrict__ Wt) {
    int idx = blockIdx.x * 256 + threadIdx.x;   // 65536 threads
    int n = idx >> 8, k = idx & 255;
    Wt[n * 256 + k] = (bf16_t)W[k * 256 + n];
}

// ---------------- atom encoder: h[n][d] = sum_f atom_emb[f][x[n][f]][d] ----
__global__ __launch_bounds__(256)
void k_atom_encoder(const int* __restrict__ xf, const float* __restrict__ aemb,
                    float* __restrict__ h) {
    int idx = blockIdx.x * 256 + threadIdx.x;
    int n = idx >> 8, d = idx & 255;
    float s = 0.f;
#pragma unroll
    for (int f = 0; f < 9; ++f) {
        int v = xf[n * 9 + f];
        s += aemb[(f * 64 + v) * 256 + d];
    }
    h[idx] = s;
}

__global__ __launch_bounds__(256)
void k_init_vn(const float* __restrict__ vemb, float* __restrict__ vn) {
    int idx = blockIdx.x * 256 + threadIdx.x;   // BB*DD
    vn[idx] = vemb[idx & 255];
}

// ---------------- hh = h + vn[batch], agg = 0 ----------------
__global__ __launch_bounds__(256)
void k_hh_zero(const float* __restrict__ h, const float* __restrict__ vn,
               const int* __restrict__ batch, float* __restrict__ hh,
               float* __restrict__ agg) {
    int idx = blockIdx.x * 256 + threadIdx.x;
    int n = idx >> 8, d = idx & 255;
    hh[idx] = h[idx] + vn[batch[n] * 256 + d];
    agg[idx] = 0.f;
}

// ---------------- edge message + scatter-add ----------------
// msg = relu(hh[src] + bond_emb_l[0][a0] + [1][a1] + [2][a2]); agg[dst] += msg
__global__ __launch_bounds__(256)
void k_edge_scatter(const float* __restrict__ hh, const int* __restrict__ eattr,
                    const int* __restrict__ src, const int* __restrict__ dst,
                    const float* __restrict__ bemb_l, float* __restrict__ agg) {
    int tid = threadIdx.x;
    int e = blockIdx.x * 4 + (tid >> 6);   // 4 edges per 256-thread block
    int d = (tid & 63) * 4;
    int s = src[e], t = dst[e];
    int a0 = eattr[e * 3 + 0], a1 = eattr[e * 3 + 1], a2 = eattr[e * 3 + 2];
    const float4 hv = *(const float4*)(hh + s * 256 + d);
    const float4 b0 = *(const float4*)(bemb_l + (0 * VBSZ + a0) * 256 + d);
    const float4 b1 = *(const float4*)(bemb_l + (1 * VBSZ + a1) * 256 + d);
    const float4 b2 = *(const float4*)(bemb_l + (2 * VBSZ + a2) * 256 + d);
    float m0 = fmaxf(hv.x + b0.x + b1.x + b2.x, 0.f);
    float m1 = fmaxf(hv.y + b0.y + b1.y + b2.y, 0.f);
    float m2 = fmaxf(hv.z + b0.z + b1.z + b2.z, 0.f);
    float m3 = fmaxf(hv.w + b0.w + b1.w + b2.w, 0.f);
    float* out = agg + t * 256 + d;
    __hip_atomic_fetch_add(out + 0, m0, __ATOMIC_RELAXED, __HIP_MEMORY_SCOPE_AGENT);
    __hip_atomic_fetch_add(out + 1, m1, __ATOMIC_RELAXED, __HIP_MEMORY_SCOPE_AGENT);
    __hip_atomic_fetch_add(out + 2, m2, __ATOMIC_RELAXED, __HIP_MEMORY_SCOPE_AGENT);
    __hip_atomic_fetch_add(out + 3, m3, __ATOMIC_RELAXED, __HIP_MEMORY_SCOPE_AGENT);
}

// ---------------- zA = bf16((1+eps)*hh + agg) ----------------
__global__ __launch_bounds__(256)
void k_make_zA(const float* __restrict__ hh, const float* __restrict__ agg,
               const float* __restrict__ eps_l, bf16_t* __restrict__ zA) {
    int idx = blockIdx.x * 256 + threadIdx.x;
    float e = 1.f + eps_l[0];
    zA[idx] = (bf16_t)(e * hh[idx] + agg[idx]);
}

// ---------------- WMMA GEMM: out = act(bn(A[M,256] @ Wt^T + bias)) ----------
// A bf16 row-major [M][256]; Wt bf16 [256][256] pre-transposed so Wt[n][k].
// The full weight matrix is staged into LDS (128 KB of the WGP's 320 KB)
// once per block — via the Tensor Data Mover when available (one descriptor-
// driven DMA issued by wave 0, tracked by TENSORcnt), else a cooperative
// load/ds_store copy. B fragments then come from ds_load.
// block = 128 threads = 4 waves; each wave owns a 16-row strip; block covers
// 64 rows; wave sweeps 16 N-tiles, 8 K-steps each -> 128 v_wmma per wave.
__global__ __launch_bounds__(128)
void k_wmma_gemm(const bf16_t* __restrict__ A, const bf16_t* __restrict__ Wt,
                 const float* __restrict__ bias,
                 const float* __restrict__ bn_g, const float* __restrict__ bn_b,
                 const float* __restrict__ bn_m, const float* __restrict__ bn_v,
                 int use_bn, int use_relu,
                 float* __restrict__ outF, bf16_t* __restrict__ outB) {
    extern __shared__ bf16_t ldsW[];   // 65536 bf16 = 128 KB (dynamic LDS)

#if HAVE_TDM
    // --- TDM stage: one 128 KB contiguous tile, global -> LDS ---
    if (threadIdx.x < 32) {            // wave 0 issues; TDM ignores EXEC
        unsigned long long ga = (unsigned long long)(uintptr_t)Wt;
        unsigned int lds_base = (unsigned int)(uintptr_t)ldsW; // LDS addr[31:0]
        // D# group 0: count=1 | lds_addr | global_addr[56:0] | type=2
        u32x4 g0;
        g0[0] = 1u;
        g0[1] = lds_base;
        g0[2] = (unsigned int)ga;
        g0[3] = ((unsigned int)(ga >> 32) & 0x01FFFFFFu) | (2u << 30);
        // D# group 1: data_size=8B; tensor_dim0=tile_dim0=16384 elems;
        // tensor_dim1=tile_dim1=1; dim0_stride=16384; no pad/iterate/multicast
        i32x8 g1;
        g1[0] = (3 << 16);                 // data_size = 3 (8 bytes)
        g1[1] = (int)(16384u << 16);       // tensor_dim0[15:0] in bits 63:48
        g1[2] = (1 << 16);                 // tensor_dim0[31:16]=0, tensor_dim1 lo=1
        g1[3] = (int)(16384u << 16);       // tensor_dim1 hi=0, tile_dim0=16384
        g1[4] = 1;                         // tile_dim1=1, tile_dim2=0
        g1[5] = 16384;                     // tensor_dim0_stride lo
        g1[6] = 0;                         // stride hi, dim1_stride lo
        g1[7] = 0;
        i32x4 z4 = {0, 0, 0, 0};           // groups 2/3 unused (<=2D tensor)
        i32x8 z8 = {0, 0, 0, 0, 0, 0, 0, 0};
        __builtin_amdgcn_tensor_load_to_lds(g0, g1, z4, z4, z8, 0);
        __builtin_amdgcn_s_wait_tensorcnt((short)0);
    }
    __syncthreads();
#else
    // --- fallback: cooperative stage, 16B per thread per step ---
    {
        const float4* gsrc = (const float4*)Wt;   // 8192 x 16B
        float4* ldst = (float4*)ldsW;
        for (int i = threadIdx.x; i < 8192; i += 128)
            ldst[i] = gsrc[i];
    }
    __syncthreads();
#endif

    int wave = threadIdx.x >> 5;
    int lane = threadIdx.x & 31;
    int m  = lane & 15;
    int kh = lane >> 4;
    int rowBase = (blockIdx.x * 4 + wave) * 16;

    // Preload full-K A fragments for this wave's 16 rows.
    // A k order used here (contiguous): lanes 0-15 hold K[0..15] of row m;
    // lanes 16-31 hold K[16..31].
    const bf16_t* arow = A + (size_t)(rowBase + m) * 256;
    v16bf a[8];
#pragma unroll
    for (int kt = 0; kt < 8; ++kt) {
        v8bf lo = *(const v8bf*)(arow + kt * 32 + kh * 16);
        v8bf hi = *(const v8bf*)(arow + kt * 32 + kh * 16 + 8);
#pragma unroll
        for (int i = 0; i < 8; ++i) { a[kt][i] = lo[i]; a[kt][i + 8] = hi[i]; }
    }

#pragma unroll 4
    for (int nt = 0; nt < 16; ++nt) {
        // B k order used here (contiguous, as A): lane (n + 16*kh) holds B[kh*16 .. kh*16+15][n]
        // = Wt[n][kt*32 + kh*16 .. +15], a contiguous 32B LDS read.
        const bf16_t* wrow = ldsW + (nt * 16 + m) * 256;
        // Batch all 8 B fragments first so the 8 WMMAs issue back-to-back
        // behind partial DS waits instead of a full wait per K-step.
        v16bf b[8];
#pragma unroll
        for (int kt = 0; kt < 8; ++kt)
            b[kt] = *(const v16bf*)(wrow + kt * 32 + kh * 16);

        v8f acc = {0.f, 0.f, 0.f, 0.f, 0.f, 0.f, 0.f, 0.f};
#pragma unroll
        for (int kt = 0; kt < 8; ++kt)
            acc = __builtin_amdgcn_wmma_f32_16x16x32_bf16(
                false, a[kt], false, b[kt], (short)0, acc, false, false);

        // epilogue: per-lane column is fixed across all 8 acc rows
        int col = nt * 16 + m;
        float bia = bias[col];
        float sc = 1.f, sh = 0.f;
        if (use_bn) {
            float g = bn_g[col], be = bn_b[col], mm = bn_m[col], vv = bn_v[col];
            sc = g * rsqrtf(vv + 1e-5f);
            sh = be - mm * sc;
        }
#pragma unroll
        for (int r = 0; r < 8; ++r) {
            int row = rowBase + r + kh * 8;
            float x = (acc[r] + bia) * sc + sh;
            if (use_relu) x = fmaxf(x, 0.f);
            size_t o = (size_t)row * 256 + col;
            if (outF) outF[o] = x;
            if (outB) outB[o] = (bf16_t)x;
        }
    }
}

// ---------------- per-graph pool: pool_bf[g][d] = bf16(vn[g][d] + sum h) ----
__global__ __launch_bounds__(256)
void k_pool(const float* __restrict__ h, const float* __restrict__ vn,
            bf16_t* __restrict__ pool_bf) {
    int g = blockIdx.x, d = threadIdx.x;
    float s = vn[g * 256 + d];
    const float* hp = h + (size_t)g * NPG * 256 + d;
    for (int i = 0; i < NPG; ++i) s += hp[i * 256];
    pool_bf[g * 256 + d] = (bf16_t)s;
}

// ---------------- driver ----------------
extern "C" void kernel_launch(void* const* d_in, const int* in_sizes, int n_in,
                              void* d_out, int out_size, void* d_ws, size_t ws_size,
                              hipStream_t stream) {
    const int*   x_feat    = (const int*)d_in[0];
    const int*   edge_attr = (const int*)d_in[1];
    const int*   src       = (const int*)d_in[2];
    const int*   dst       = (const int*)d_in[3];
    const int*   batch     = (const int*)d_in[4];
    const float* atom_emb  = (const float*)d_in[6];
    const float* bond_emb  = (const float*)d_in[7];
    const float* vn_emb    = (const float*)d_in[8];
    const float* eps       = (const float*)d_in[9];
    const float* gin_w1    = (const float*)d_in[10];
    const float* gin_b1    = (const float*)d_in[11];
    const float* gin_bn_g  = (const float*)d_in[12];
    const float* gin_bn_b  = (const float*)d_in[13];
    const float* gin_bn_m  = (const float*)d_in[14];
    const float* gin_bn_v  = (const float*)d_in[15];
    const float* gin_w2    = (const float*)d_in[16];
    const float* gin_b2    = (const float*)d_in[17];
    const float* out_bn_g  = (const float*)d_in[18];
    const float* out_bn_b  = (const float*)d_in[19];
    const float* out_bn_m  = (const float*)d_in[20];
    const float* out_bn_v  = (const float*)d_in[21];
    const float* vn_w1     = (const float*)d_in[22];
    const float* vn_b1     = (const float*)d_in[23];
    const float* vn_bn1_g  = (const float*)d_in[24];
    const float* vn_bn1_b  = (const float*)d_in[25];
    const float* vn_bn1_m  = (const float*)d_in[26];
    const float* vn_bn1_v  = (const float*)d_in[27];
    const float* vn_w2     = (const float*)d_in[28];
    const float* vn_b2     = (const float*)d_in[29];
    const float* vn_bn2_g  = (const float*)d_in[30];
    const float* vn_bn2_b  = (const float*)d_in[31];
    const float* vn_bn2_m  = (const float*)d_in[32];
    const float* vn_bn2_v  = (const float*)d_in[33];

    float* h = (float*)d_out;                 // h lives in d_out (N*D f32)

    // workspace carve-out (~27 MB total)
    char* ws = (char*)d_ws;
    size_t off = 0;
    auto carve = [&](size_t bytes) -> void* {
        void* p = ws + off;
        off = (off + bytes + 255) & ~(size_t)255;
        return p;
    };
    float*  hh   = (float*)carve((size_t)NN * DD * 4);
    float*  agg  = (float*)carve((size_t)NN * DD * 4);
    bf16_t* zA   = (bf16_t*)carve((size_t)NN * DD * 2);
    bf16_t* z1   = (bf16_t*)carve((size_t)NN * DD * 2);
    float*  vn   = (float*)carve((size_t)BB * DD * 4);
    bf16_t* pool = (bf16_t*)carve((size_t)BB * DD * 2);
    bf16_t* vt1  = (bf16_t*)carve((size_t)BB * DD * 2);
    bf16_t* WtG1[LL], *WtG2[LL];
    for (int l = 0; l < LL; ++l) {
        WtG1[l] = (bf16_t*)carve(256 * 256 * 2);
        WtG2[l] = (bf16_t*)carve(256 * 256 * 2);
    }
    bf16_t* WtV1 = (bf16_t*)carve(256 * 256 * 2);
    bf16_t* WtV2 = (bf16_t*)carve(256 * 256 * 2);

    // ---- prep: transpose+convert weights, atom encoder, init vn ----
    for (int l = 0; l < LL; ++l) {
        k_transpose_bf16<<<256, 256, 0, stream>>>(gin_w1 + (size_t)l * 65536, WtG1[l]);
        k_transpose_bf16<<<256, 256, 0, stream>>>(gin_w2 + (size_t)l * 65536, WtG2[l]);
    }
    k_transpose_bf16<<<256, 256, 0, stream>>>(vn_w1, WtV1);
    k_transpose_bf16<<<256, 256, 0, stream>>>(vn_w2, WtV2);
    k_atom_encoder<<<(NN * DD) / 256, 256, 0, stream>>>(x_feat, atom_emb, h);
    k_init_vn<<<(BB * DD) / 256, 256, 0, stream>>>(vn_emb, vn);

    // ---- main loops ----
    for (int l = 0; l < LL; ++l) {
        const float* bemb_l = bond_emb + (size_t)l * 3 * VBSZ * DD;
        for (int it = 0; it < NIT; ++it) {
            k_hh_zero<<<(NN * DD) / 256, 256, 0, stream>>>(h, vn, batch, hh, agg);
            k_edge_scatter<<<EE / 4, 256, 0, stream>>>(hh, edge_attr, src, dst, bemb_l, agg);
            k_make_zA<<<(NN * DD) / 256, 256, 0, stream>>>(hh, agg, eps + l, zA);
            // z1 = relu(BN(zA @ W1 + b1)) -> bf16
            k_wmma_gemm<<<NN / 64, 128, LDS_W_BYTES, stream>>>(
                zA, WtG1[l], gin_b1 + l * 256,
                gin_bn_g + l * 256, gin_bn_b + l * 256, gin_bn_m + l * 256, gin_bn_v + l * 256,
                1, 1, nullptr, z1);
            // h = act(bn?(z1 @ W2 + b2)) -> f32 (in d_out)
            int last = (it == NIT - 1);
            int use_bn2 = last;
            int use_relu2 = !(l == LL - 1 && last);
            k_wmma_gemm<<<NN / 64, 128, LDS_W_BYTES, stream>>>(
                z1, WtG2[l], gin_b2 + l * 256,
                out_bn_g + l * 256, out_bn_b + l * 256, out_bn_m + l * 256, out_bn_v + l * 256,
                use_bn2, use_relu2, h, nullptr);
        }
        if (l < LL - 1) {
            // virtual node update
            k_pool<<<BB, 256, 0, stream>>>(h, vn, pool);
            k_wmma_gemm<<<BB / 64, 128, LDS_W_BYTES, stream>>>(
                pool, WtV1, vn_b1,
                vn_bn1_g, vn_bn1_b, vn_bn1_m, vn_bn1_v,
                1, 1, nullptr, vt1);
            k_wmma_gemm<<<BB / 64, 128, LDS_W_BYTES, stream>>>(
                vt1, WtV2, vn_b2,
                vn_bn2_g, vn_bn2_b, vn_bn2_m, vn_bn2_v,
                1, 1, vn, nullptr);
        }
    }
}
